// MultiheadAttention_2310692405797
// MI455X (gfx1250) — compile-verified, matrix operand order checked by reading, not run on hardware
//
#include <hip/hip_runtime.h>
#include <hip/hip_bf16.h>

// MultiheadAttention for MI455X (gfx1250, wave32, WMMA + TDM).
// E=1024, H=16, D=64, B=1, L=S=4096.
// d_out = [out (4096x1024 f32)] ++ [attn (16x4096x4096 f32)]
// Workspace: Qh(8MB f16) | Kh(8MB f16) | Vt(8MB f16, [E][L]) | O(8MB f16) = 32 MB.

typedef __attribute__((ext_vector_type(16))) _Float16 v16h;
typedef __attribute__((ext_vector_type(8)))  _Float16 v8h;
typedef __attribute__((ext_vector_type(8)))  float    v8f;
typedef __attribute__((ext_vector_type(4)))  float    v4f;
typedef __attribute__((ext_vector_type(4)))  unsigned int u32x4;
typedef __attribute__((ext_vector_type(8)))  unsigned int u32x8;

#define E_DIM 1024
#define L_DIM 4096
#define H_DIM 16
#define D_DIM 64

// ---- WMMA f32 = f16 x f16 + f32 -------------------------------------------
__device__ __forceinline__ v8f wmma_f16(v16h a, v16h b, v8f c) {
  return __builtin_amdgcn_wmma_f32_16x16x32_f16(false, a, false, b, (short)0, c,
                                                false, false);
}

// A/B fragment for V_WMMA_F32_16X16X32_F16 (ISA 7.12.2):
// lane l holds row (l&15); K-chunks [ko..ko+7] in v0..v3 and [ko+16..ko+23] in
// v4..v7 with ko = 8*(l>>4).  Two b128 loads from a row-major f16 matrix.
__device__ __forceinline__ v16h load_frag_h(const _Float16* base, int ld, int lane) {
  const int r  = lane & 15;
  const int ko = (lane >> 4) << 3;
  const _Float16* p = base + (long)r * ld + ko;
  v8h lo = *(const v8h*)p;
  v8h hi = *(const v8h*)(p + 16);
  return __builtin_shufflevector(lo, hi, 0, 1, 2, 3, 4, 5, 6, 7,
                                 8, 9, 10, 11, 12, 13, 14, 15);
}

// Same fragment sourced from an f32 row-major matrix: 4x b128 loads + cvt.
__device__ __forceinline__ v16h load_frag_cvt(const float* base, int ld, int lane) {
  const int r  = lane & 15;
  const int ko = (lane >> 4) << 3;
  const float* p = base + (long)r * ld + ko;
  v4f a0 = *(const v4f*)p;
  v4f a1 = *(const v4f*)(p + 4);
  v4f b0 = *(const v4f*)(p + 16);
  v4f b1 = *(const v4f*)(p + 20);
  v16h f;
#pragma unroll
  for (int i = 0; i < 4; ++i) {
    f[i]      = (_Float16)a0[i];
    f[4 + i]  = (_Float16)a1[i];
    f[8 + i]  = (_Float16)b0[i];
    f[12 + i] = (_Float16)b1[i];
  }
  return f;
}

// ---------------------------------------------------------------------------
// K1: fused QKV projection.  z selects (X,W,b):  out = X @ W^T + b  (f16 out)
//     z==2 (V) stores transposed: Vt[e][l].
// Block 256 thr = 8 waves; block tile 128x128; wave tile 32x64 (2x4 WMMAs).
// ---------------------------------------------------------------------------
__global__ __launch_bounds__(256) void proj_qkv_k(
    const float* __restrict__ qin, const float* __restrict__ kin,
    const float* __restrict__ vin, const float* __restrict__ wq,
    const float* __restrict__ wk, const float* __restrict__ wv,
    const float* __restrict__ bq, const float* __restrict__ bk,
    const float* __restrict__ bv, _Float16* __restrict__ Qh,
    _Float16* __restrict__ Kh, _Float16* __restrict__ Vt) {
  const int z = blockIdx.z;
  const float *X, *W, *bias;
  if (z == 0)      { X = qin; W = wq; bias = bq; }
  else if (z == 1) { X = kin; W = wk; bias = bk; }
  else             { X = vin; W = wv; bias = bv; }

  const int lane = threadIdx.x & 31;
  const int wave = threadIdx.x >> 5;
  const int row0 = blockIdx.y * 128 + (wave & 3) * 32;
  const int col0 = blockIdx.x * 128 + (wave >> 2) * 64;

  v8f acc[2][4] = {};
  for (int k0 = 0; k0 < E_DIM; k0 += 32) {
    v16h a[2], b[4];
#pragma unroll
    for (int i = 0; i < 2; ++i)
      a[i] = load_frag_cvt(X + (long)(row0 + 16 * i) * E_DIM + k0, E_DIM, lane);
#pragma unroll
    for (int j = 0; j < 4; ++j)
      b[j] = load_frag_cvt(W + (long)(col0 + 16 * j) * E_DIM + k0, E_DIM, lane);
#pragma unroll
    for (int i = 0; i < 2; ++i)
#pragma unroll
      for (int j = 0; j < 4; ++j)
        acc[i][j] = wmma_f16(a[i], b[j], acc[i][j]);
  }

  const int n  = lane & 15;
  const int mb = (lane >> 4) * 8;
  // Hoisted z-branch: straight-line store bodies (no per-element branch).
  if (z == 2) {
#pragma unroll
    for (int i = 0; i < 2; ++i)
#pragma unroll
      for (int j = 0; j < 4; ++j) {
        const int c = col0 + 16 * j + n;
        const float bb = bias[c];
#pragma unroll
        for (int v = 0; v < 8; ++v) {
          const int r = row0 + 16 * i + mb + v;
          Vt[(long)c * L_DIM + r] = (_Float16)(acc[i][j][v] + bb);
        }
      }
  } else {
    _Float16* __restrict__ dst = (z == 1) ? Kh : Qh;
#pragma unroll
    for (int i = 0; i < 2; ++i)
#pragma unroll
      for (int j = 0; j < 4; ++j) {
        const int c = col0 + 16 * j + n;
        const float bb = bias[c];
#pragma unroll
        for (int v = 0; v < 8; ++v) {
          const int r = row0 + 16 * i + mb + v;
          dst[(long)r * E_DIM + c] = (_Float16)(acc[i][j][v] + bb);
        }
      }
  }
}

// ---------------------------------------------------------------------------
// K2: per-head scores = (Q_h @ K_h^T) / sqrt(D), optional causal mask.
// K=64 -> 2 WMMA k-steps.  f32 out into d_out attn region.
// ---------------------------------------------------------------------------
__global__ __launch_bounds__(256) void scores_k(
    const _Float16* __restrict__ Qh, const _Float16* __restrict__ Kh,
    float* __restrict__ attnW, const int* __restrict__ iscausal) {
  const int h    = blockIdx.z;
  const int lane = threadIdx.x & 31;
  const int wave = threadIdx.x >> 5;
  const int row0 = blockIdx.y * 128 + (wave & 3) * 32;
  const int col0 = blockIdx.x * 128 + (wave >> 2) * 64;
  const int causal = iscausal[0];

  const _Float16* Ab = Qh + h * D_DIM;
  const _Float16* Bb = Kh + h * D_DIM;

  v8f acc[2][4] = {};
#pragma unroll
  for (int k0 = 0; k0 < D_DIM; k0 += 32) {
    v16h a[2], b[4];
#pragma unroll
    for (int i = 0; i < 2; ++i)
      a[i] = load_frag_h(Ab + (long)(row0 + 16 * i) * E_DIM + k0, E_DIM, lane);
#pragma unroll
    for (int j = 0; j < 4; ++j)
      b[j] = load_frag_h(Bb + (long)(col0 + 16 * j) * E_DIM + k0, E_DIM, lane);
#pragma unroll
    for (int i = 0; i < 2; ++i)
#pragma unroll
      for (int j = 0; j < 4; ++j)
        acc[i][j] = wmma_f16(a[i], b[j], acc[i][j]);
  }

  float* out = attnW + (long)h * L_DIM * L_DIM;
  const int n  = lane & 15;
  const int mb = (lane >> 4) * 8;
#pragma unroll
  for (int i = 0; i < 2; ++i)
#pragma unroll
    for (int j = 0; j < 4; ++j) {
      const int c = col0 + 16 * j + n;
#pragma unroll
      for (int v = 0; v < 8; ++v) {
        const int r = row0 + 16 * i + mb + v;
        float val = acc[i][j][v] * 0.125f;              // 1/sqrt(64)
        if (causal && (c > r)) val = -__builtin_inff();
        out[(long)r * L_DIM + c] = val;
      }
    }
}

// ---------------------------------------------------------------------------
// K3: in-place row softmax over attnW; one 256-thread block per 4096-row.
// ---------------------------------------------------------------------------
__global__ __launch_bounds__(256) void softmax_k(float* __restrict__ attnW) {
  float* p = attnW + (long)blockIdx.x * L_DIM;
  __shared__ float red[256];
  const int t = threadIdx.x;
  float4* pv = (float4*)p;

  float4 x[4];
  float m = -3.4e38f;
#pragma unroll
  for (int i = 0; i < 4; ++i) {
    x[i] = pv[t * 4 + i];
    m = fmaxf(m, fmaxf(fmaxf(x[i].x, x[i].y), fmaxf(x[i].z, x[i].w)));
  }
  red[t] = m;
  __syncthreads();
  for (int s = 128; s > 0; s >>= 1) {
    if (t < s) red[t] = fmaxf(red[t], red[t + s]);
    __syncthreads();
  }
  const float rm = red[0];
  __syncthreads();

  float sum = 0.f;
#pragma unroll
  for (int i = 0; i < 4; ++i) {
    x[i].x = __expf(x[i].x - rm);
    x[i].y = __expf(x[i].y - rm);
    x[i].z = __expf(x[i].z - rm);
    x[i].w = __expf(x[i].w - rm);
    sum += x[i].x + x[i].y + x[i].z + x[i].w;
  }
  red[t] = sum;
  __syncthreads();
  for (int s = 128; s > 0; s >>= 1) {
    if (t < s) red[t] += red[t + s];
    __syncthreads();
  }
  const float inv = 1.0f / red[0];
#pragma unroll
  for (int i = 0; i < 4; ++i) {
    x[i].x *= inv; x[i].y *= inv; x[i].z *= inv; x[i].w *= inv;
    pv[t * 4 + i] = x[i];
  }
}

// ---------------------------------------------------------------------------
// K4: per-head attn_out = P @ V_h.
// Block tile 256x64: 8 waves, each 32 rows x full 64 cols (2x4 WMMAs) so each
// attnW element (the 1 GB stream) is read exactly once.
// The block-uniform B tile (64 rows of Vt x 32 halfs = 4 KB) is staged into
// LDS by the Tensor Data Mover, DOUBLE-BUFFERED: at loop top wave0 waits for
// tile i (the only outstanding TDM op), barrier, then issues tile i+1 into
// the other buffer while all waves compute on tile i.
// ---------------------------------------------------------------------------
__global__ __launch_bounds__(256) void pv_k(
    const float* __restrict__ attnW, const _Float16* __restrict__ Vt,
    _Float16* __restrict__ O) {
  __shared__ alignas(16) _Float16 btile[2][D_DIM * 32];  // 2 x 4 KB

  const int h    = blockIdx.z;
  const int lane = threadIdx.x & 31;
  const int wave = threadIdx.x >> 5;
  const int row0 = blockIdx.y * 256 + wave * 32;

  const float* Ab = attnW + (long)h * L_DIM * L_DIM;
  const unsigned long long vbase =
      (unsigned long long)(const void*)(Vt + (long)h * D_DIM * L_DIM);
  const unsigned ldsoff[2] = {
      (unsigned)(__SIZE_TYPE__)(&btile[0][0]),   // LDS byte offsets
      (unsigned)(__SIZE_TYPE__)(&btile[1][0])};

  // ---- D# group1 (256b, loop-invariant): data_size=2B; tensor_dim0=4096,
  // tensor_dim1=64; tile_dim0=32, tile_dim1=64; tensor_dim0_stride=4096.
  const u32x8 g1 = {
      1u << 16,                                  // [17:16] data_size = 1 (2B)
      (4096u & 0xFFFFu) << 16,                   // [79:48] tensor_dim0 lo16
      (4096u >> 16) | (64u << 16),               // tensor_dim0 hi | tensor_dim1 lo
      (32u << 16),                               // tensor_dim1 hi | [127:112] tile_dim0
      64u,                                       // [143:128] tile_dim1, tile_dim2=0
      4096u,                                     // [207:160] tensor_dim0_stride lo32
      0u, 0u};

  auto issue_tdm = [&](int k0, unsigned lds) {
    const unsigned long long ga = vbase + (unsigned long long)k0 * 2ull;
    const u32x4 g0 = {
        1u,                                             // count=1 (valid D#)
        lds,                                            // [63:32] lds_addr
        (unsigned)ga,                                   // global_addr lo32
        ((unsigned)(ga >> 32) & 0x1FFFFFFu) | (2u << 30)};  // ga hi | type=2
    asm volatile("tensor_load_to_lds %0, %1" ::"s"(g0), "s"(g1) : "memory");
  };

  if (threadIdx.x < 32) issue_tdm(0, ldsoff[0]);       // prologue: tile 0

  v8f acc[2][4] = {};
  int ib = 0;
  for (int k0 = 0; k0 < L_DIM; k0 += 32, ib ^= 1) {
    if (threadIdx.x < 32) __builtin_amdgcn_s_wait_tensorcnt(0);  // tile i done
    __syncthreads();                                   // visible to all waves
    if (threadIdx.x < 32 && (k0 + 32) < L_DIM)
      issue_tdm(k0 + 32, ldsoff[ib ^ 1]);              // overlap tile i+1

    __builtin_prefetch(Ab + (long)row0 * L_DIM + k0 + 64, 0, 3);
    v16h a[2], b[4];
#pragma unroll
    for (int i = 0; i < 2; ++i)
      a[i] = load_frag_cvt(Ab + (long)(row0 + 16 * i) * L_DIM + k0, L_DIM, lane);
#pragma unroll
    for (int j = 0; j < 4; ++j)
      b[j] = load_frag_h(&btile[ib][16 * j * 32], 32, lane);  // ds_load_b128 x2
#pragma unroll
    for (int i = 0; i < 2; ++i)
#pragma unroll
      for (int j = 0; j < 4; ++j)
        acc[i][j] = wmma_f16(a[i], b[j], acc[i][j]);
  }

  const int n  = lane & 15;
  const int mb = (lane >> 4) * 8;
#pragma unroll
  for (int i = 0; i < 2; ++i)
#pragma unroll
    for (int j = 0; j < 4; ++j) {
      const int c = h * D_DIM + 16 * j + n;
#pragma unroll
      for (int v = 0; v < 8; ++v) {
        const int r = row0 + 16 * i + mb + v;
        O[(long)r * E_DIM + c] = (_Float16)acc[i][j][v];
      }
    }
}

// ---------------------------------------------------------------------------
// K5: out = O @ fc_w^T + fc_b  (f32 out).  Same tiling as K1.
// ---------------------------------------------------------------------------
__global__ __launch_bounds__(256) void outproj_k(
    const _Float16* __restrict__ O, const float* __restrict__ fcw,
    const float* __restrict__ fcb, float* __restrict__ out) {
  const int lane = threadIdx.x & 31;
  const int wave = threadIdx.x >> 5;
  const int row0 = blockIdx.y * 128 + (wave & 3) * 32;
  const int col0 = blockIdx.x * 128 + (wave >> 2) * 64;

  v8f acc[2][4] = {};
  for (int k0 = 0; k0 < E_DIM; k0 += 32) {
    v16h a[2], b[4];
#pragma unroll
    for (int i = 0; i < 2; ++i)
      a[i] = load_frag_h(O + (long)(row0 + 16 * i) * E_DIM + k0, E_DIM, lane);
#pragma unroll
    for (int j = 0; j < 4; ++j)
      b[j] = load_frag_cvt(fcw + (long)(col0 + 16 * j) * E_DIM + k0, E_DIM, lane);
#pragma unroll
    for (int i = 0; i < 2; ++i)
#pragma unroll
      for (int j = 0; j < 4; ++j)
        acc[i][j] = wmma_f16(a[i], b[j], acc[i][j]);
  }

  const int n  = lane & 15;
  const int mb = (lane >> 4) * 8;
#pragma unroll
  for (int i = 0; i < 2; ++i)
#pragma unroll
    for (int j = 0; j < 4; ++j) {
      const int c = col0 + 16 * j + n;
      const float bb = fcb[c];
#pragma unroll
      for (int v = 0; v < 8; ++v) {
        const int r = row0 + 16 * i + mb + v;
        out[(long)r * E_DIM + c] = acc[i][j][v] + bb;
      }
    }
}

// ---------------------------------------------------------------------------
extern "C" void kernel_launch(void* const* d_in, const int* in_sizes, int n_in,
                              void* d_out, int out_size, void* d_ws,
                              size_t ws_size, hipStream_t stream) {
  const float* query = (const float*)d_in[0];
  const float* key   = (const float*)d_in[1];
  const float* value = (const float*)d_in[2];
  const float* wq_w  = (const float*)d_in[3];
  const float* wq_b  = (const float*)d_in[4];
  const float* wk_w  = (const float*)d_in[5];
  const float* wk_b  = (const float*)d_in[6];
  const float* wv_w  = (const float*)d_in[7];
  const float* wv_b  = (const float*)d_in[8];
  const float* fc_w  = (const float*)d_in[9];
  const float* fc_b  = (const float*)d_in[10];
  const int* iscausal = (const int*)d_in[11];

  float* out   = (float*)d_out;
  float* attnW = out + (long)L_DIM * E_DIM;  // 16x4096x4096 softmax output

  // Workspace layout (f16): Qh | Kh | Vt(transposed [E][L]) | O  = 32 MB total
  _Float16* Qh = (_Float16*)d_ws;
  _Float16* Kh = Qh + (long)L_DIM * E_DIM;
  _Float16* Vt = Kh + (long)L_DIM * E_DIM;
  _Float16* O  = Vt + (long)L_DIM * E_DIM;

  const dim3 blk(256);
  proj_qkv_k<<<dim3(E_DIM / 128, L_DIM / 128, 3), blk, 0, stream>>>(
      query, key, value, wq_w, wk_w, wv_w, wq_b, wk_b, wv_b, Qh, Kh, Vt);
  scores_k<<<dim3(L_DIM / 128, L_DIM / 128, H_DIM), blk, 0, stream>>>(
      Qh, Kh, attnW, iscausal);
  softmax_k<<<dim3(H_DIM * L_DIM), blk, 0, stream>>>(attnW);
  pv_k<<<dim3(1, L_DIM / 256, H_DIM), blk, 0, stream>>>(attnW, Vt, O);
  outproj_k<<<dim3(E_DIM / 128, L_DIM / 128, 1), blk, 0, stream>>>(
      O, fc_w, fc_b, out);
}
